// GCN_MLP_2774548873728
// MI455X (gfx1250) — matrix operand order repaired and checked by reading, not compiled, not run on hardware
//
#include <hip/hip_runtime.h>
#include <hip/hip_bf16.h>
#include <stdint.h>

// ---------------------------------------------------------------------------
// Problem constants (fixed by the reference)
// ---------------------------------------------------------------------------
#define N_NODES 10000
#define N_EDGES 160000
#define N_LIG   5000
#define B_PAIRS 16384
#define D_IN    2813
#define D_IN_P  2816      // padded to multiple of 32
#define D_H1    1024
#define D_H2    400
#define D_FEAT  800       // 2*D_H2, multiple of 32

typedef __attribute__((ext_vector_type(16))) __bf16 v16bf;
typedef __attribute__((ext_vector_type(8)))  __bf16 v8bf;
typedef __attribute__((ext_vector_type(8)))  float  v8f;

__device__ __forceinline__ unsigned short f2bf_raw(float x) {
  // round-to-nearest-even f32 -> bf16
  union { float f; unsigned u; } c; c.f = x;
  unsigned r = c.u + 0x7FFFu + ((c.u >> 16) & 1u);
  return (unsigned short)(r >> 16);
}

__device__ __forceinline__ int imin(int a, int b) { return a < b ? a : b; }

// ---------------------------------------------------------------------------
// bf16 WMMA GEMM:  C[M,N] = epi( A[M,Kp] @ Bt[N,Kp]^T + bias + residual )
// A  : bf16, M rows x Kp (Kp % 32 == 0, rows 16B-aligned)
// Bt : bf16, N rows x Kp (weights pre-transposed)
// BM=BN=128, BK=32; 256 threads = 8 waves; each wave owns a 32x64 patch
// (2x4 of 16x16 WMMA tiles). Double-buffered LDS filled by CDNA5
// global_load_async_to_lds_b128 (ASYNCcnt path, no staging VGPRs).
// ---------------------------------------------------------------------------
#define BM 128
#define BN 128
#define BK 32
#define LDSS 40            // bf16 elems per LDS row; 80B stride, 16B aligned
#define LDS_BUF_BYTES (BM * LDSS * 2)   // 10240B per buffer (same for A and B)

__global__ __launch_bounds__(256) void wmma_gemm_bf16(
    const unsigned short* __restrict__ A, const unsigned short* __restrict__ Bt,
    const float* __restrict__ bias, const float* __restrict__ residual,
    float* __restrict__ C, int M, int N, int Kp, int epi /*0 none, 1 relu*/)
{
  __shared__ __align__(16) unsigned short sA[2][BM * LDSS];
  __shared__ __align__(16) unsigned short sB[2][BN * LDSS];

  const int tid  = threadIdx.x;
  const int lane = tid & 31;
  const int wave = tid >> 5;
  const int half = lane >> 4;
  const int l16  = lane & 15;
  const int m0 = blockIdx.y * BM;
  const int n0 = blockIdx.x * BN;
  const int wm = (wave & 3) << 5;   // 4 waves along M, 32 rows each
  const int wn = (wave >> 2) << 6;  // 2 waves along N, 64 cols each
  const int nK = Kp / BK;

  v8f acc[2][4] = {};

  // ---- per-thread staging addresses, hoisted out of the K loop -----------
  // 512 16B-chunks per tile; thread handles chunks {tid, tid+256} of A and B.
  const int rowT = tid >> 2;              // 0..63
  const int chE  = (tid & 3) << 3;        // element offset: 0,8,16,24
  const unsigned short* gA0 = A  + (size_t)imin(m0 + rowT,      M - 1) * Kp + chE;
  const unsigned short* gA1 = A  + (size_t)imin(m0 + rowT + 64, M - 1) * Kp + chE;
  const unsigned short* gB0 = Bt + (size_t)imin(n0 + rowT,      N - 1) * Kp + chE;
  const unsigned short* gB1 = Bt + (size_t)imin(n0 + rowT + 64, N - 1) * Kp + chE;
  const unsigned lA0 = (unsigned)(uintptr_t)&sA[0][ rowT       * LDSS + chE];
  const unsigned lA1 = (unsigned)(uintptr_t)&sA[0][(rowT + 64) * LDSS + chE];
  const unsigned lB0 = (unsigned)(uintptr_t)&sB[0][ rowT       * LDSS + chE];
  const unsigned lB1 = (unsigned)(uintptr_t)&sB[0][(rowT + 64) * LDSS + chE];

  // async DMA of one BKx(BM|BN) double-buffer tile: 4 x 16B per thread
  auto stage = [&](int buf, int k0) {
    const unsigned boff = buf ? (unsigned)LDS_BUF_BYTES : 0u;
    asm volatile("global_load_async_to_lds_b128 %0, %1, off"
                 :: "v"(lA0 + boff), "v"(gA0 + k0) : "memory");
    asm volatile("global_load_async_to_lds_b128 %0, %1, off"
                 :: "v"(lA1 + boff), "v"(gA1 + k0) : "memory");
    asm volatile("global_load_async_to_lds_b128 %0, %1, off"
                 :: "v"(lB0 + boff), "v"(gB0 + k0) : "memory");
    asm volatile("global_load_async_to_lds_b128 %0, %1, off"
                 :: "v"(lB1 + boff), "v"(gB1 + k0) : "memory");
  };

  stage(0, 0);
  for (int kt = 0; kt < nK; ++kt) {
    // all of this wave's async LDS fills done, then block-wide visibility
    asm volatile("s_wait_asynccnt 0x0" ::: "memory");
    __syncthreads();
    const int cur = kt & 1;

    v16bf af[2], bfr[4];
#pragma unroll
    for (int t = 0; t < 2; ++t) {
      // A 16x32 layout: lane half 0 -> K {0..7,16..23}; half 1 -> {8..15,24..31}
      const unsigned short* pa = &sA[cur][(wm + t * 16 + l16) * LDSS + half * 8];
      v8bf lo = *(const v8bf*)pa;
      v8bf hi = *(const v8bf*)(pa + 16);
      af[t] = __builtin_shufflevector(lo, hi,
              0,1,2,3,4,5,6,7,8,9,10,11,12,13,14,15);
    }
#pragma unroll
    for (int u = 0; u < 4; ++u) {
      // B 32x16 layout: lane half 0 -> K {0..7,16..23}; half 1 -> {8..15,24..31}
      const unsigned short* pb = &sB[cur][(wn + u * 16 + l16) * LDSS + half * 8];
      v8bf lo = *(const v8bf*)pb;
      v8bf hi = *(const v8bf*)(pb + 16);
      bfr[u] = __builtin_shufflevector(lo, hi,
              0,1,2,3,4,5,6,7,8,9,10,11,12,13,14,15);
    }

    if (kt + 1 < nK) stage(1 - cur, (kt + 1) * BK);  // overlap with WMMAs

#pragma unroll
    for (int t = 0; t < 2; ++t)
#pragma unroll
      for (int u = 0; u < 4; ++u)
        acc[t][u] = __builtin_amdgcn_wmma_f32_16x16x32_bf16(
            false, af[t], false, bfr[u], (short)0, acc[t][u], false, false);
  }

  // epilogue: C/D layout VGPR v -> row v + half*8, col = l16
#pragma unroll
  for (int t = 0; t < 2; ++t)
#pragma unroll
    for (int u = 0; u < 4; ++u)
#pragma unroll
      for (int v = 0; v < 8; ++v) {
        int row = m0 + wm + t * 16 + v + half * 8;
        int col = n0 + wn + u * 16 + l16;
        if (row < M && col < N) {
          float r = acc[t][u][v];
          if (bias)     r += bias[col];
          if (residual) r += residual[(size_t)row * N + col];
          if (epi)      r = fmaxf(r, 0.f);
          C[(size_t)row * N + col] = r;
        }
      }
}

// ---------------------------------------------------------------------------
// Conversion / layout kernels (one-shot, bandwidth-trivial vs GEMM compute)
// ---------------------------------------------------------------------------
__global__ void convert_pad_bf16(const float* __restrict__ A,
                                 unsigned short* __restrict__ out,
                                 int M, int K, int Kp) {
  size_t i = (size_t)blockIdx.x * blockDim.x + threadIdx.x;
  size_t total = (size_t)M * Kp;
  if (i >= total) return;
  int m = (int)(i / Kp), k = (int)(i % Kp);
  out[i] = (k < K) ? f2bf_raw(A[(size_t)m * K + k]) : (unsigned short)0;
}

__global__ void transpose_convert_bf16(const float* __restrict__ B,
                                       unsigned short* __restrict__ Bt,
                                       int K, int N, int Kp) {
  size_t i = (size_t)blockIdx.x * blockDim.x + threadIdx.x;
  size_t total = (size_t)N * Kp;
  if (i >= total) return;
  int n = (int)(i / Kp), k = (int)(i % Kp);
  Bt[i] = (k < K) ? f2bf_raw(B[(size_t)k * N + n]) : (unsigned short)0;
}

__global__ void gather_feat_bf16(const float* __restrict__ H2,
                                 const int* __restrict__ pairs,
                                 unsigned short* __restrict__ feat) {
  size_t i = (size_t)blockIdx.x * blockDim.x + threadIdx.x;
  size_t total = (size_t)B_PAIRS * D_FEAT;
  if (i >= total) return;
  int p = (int)(i / D_FEAT), k = (int)(i % D_FEAT);
  int i0 = pairs[2 * p], i1 = pairs[2 * p + 1];
  float v = (k < D_H2) ? H2[(size_t)i0 * D_H2 + k]
                       : H2[(size_t)(N_LIG + i1) * D_H2 + (k - D_H2)];
  feat[i] = f2bf_raw(v);
}

// ---------------------------------------------------------------------------
// Graph aggregation helpers
// ---------------------------------------------------------------------------
__global__ void fill_f32(float* __restrict__ p, float v, int n) {
  int i = blockIdx.x * blockDim.x + threadIdx.x;
  if (i < n) p[i] = v;
}

__global__ void deg_count(const int* __restrict__ dst, float* __restrict__ deg, int nE) {
  int e = blockIdx.x * blockDim.x + threadIdx.x;
  if (e < nE) atomicAdd(&deg[dst[e]], 1.0f);
}

__global__ void rsqrt_k(const float* __restrict__ deg, float* __restrict__ dinv, int n) {
  int i = blockIdx.x * blockDim.x + threadIdx.x;
  if (i < n) {
    float d = deg[i];
    dinv[i] = d > 0.f ? rsqrtf(d) : 0.f;
  }
}

// Agg[i,f] = T[i,f] * dinv[i]^2  (self-loop term; also initializes Agg)
__global__ void self_init(const float* __restrict__ T, const float* __restrict__ dinv,
                          float* __restrict__ Agg, int n, int F) {
  size_t i = (size_t)blockIdx.x * blockDim.x + threadIdx.x;
  size_t total = (size_t)n * F;
  if (i < total) {
    int node = (int)(i / F);
    float di = dinv[node];
    Agg[i] = T[i] * di * di;
  }
}

// Agg[dst] += T[src] * dinv[src]*dinv[dst]  (one block per edge, float4 loads)
__global__ void edge_agg(const float* __restrict__ T, const int* __restrict__ src,
                         const int* __restrict__ dst, const float* __restrict__ dinv,
                         float* __restrict__ Agg, int F) {
  int e = blockIdx.x;
  int s = src[e], d = dst[e];
  float nrm = dinv[s] * dinv[d];
  const float* ts = T + (size_t)s * F;
  float* ad = Agg + (size_t)d * F;
  for (int f = threadIdx.x * 4; f < F; f += blockDim.x * 4) {
    float4 v = *(const float4*)(ts + f);
    atomicAdd(&ad[f + 0], v.x * nrm);
    atomicAdd(&ad[f + 1], v.y * nrm);
    atomicAdd(&ad[f + 2], v.z * nrm);
    atomicAdd(&ad[f + 3], v.w * nrm);
  }
}

// ---------------------------------------------------------------------------
// Final 32 -> 1 projection + sigmoid
// ---------------------------------------------------------------------------
__global__ void fc3_sigmoid(const float* __restrict__ Z2, const float* __restrict__ w,
                            const float* __restrict__ b, float* __restrict__ out, int P) {
  int p = blockIdx.x * blockDim.x + threadIdx.x;
  if (p >= P) return;
  float s = b[0];
#pragma unroll
  for (int k = 0; k < 32; ++k) s += Z2[(size_t)p * 32 + k] * w[k];
  out[p] = 1.f / (1.f + __expf(-s));
}

// ---------------------------------------------------------------------------
// Orchestration
// ---------------------------------------------------------------------------
extern "C" void kernel_launch(void* const* d_in, const int* in_sizes, int n_in,
                              void* d_out, int out_size, void* d_ws, size_t ws_size,
                              hipStream_t stream) {
  (void)in_sizes; (void)n_in; (void)out_size; (void)ws_size;

  const float* x     = (const float*)d_in[0];
  const int*   eidx  = (const int*)d_in[1];       // (2, N_EDGES)
  const int*   pairs = (const int*)d_in[2];       // (B_PAIRS, 2)
  const float* W1    = (const float*)d_in[4];
  const float* b1    = (const float*)d_in[5];
  const float* W2    = (const float*)d_in[6];
  const float* b2    = (const float*)d_in[7];
  const float* f1w   = (const float*)d_in[8];
  const float* f1b   = (const float*)d_in[9];
  const float* f2w   = (const float*)d_in[10];
  const float* f2b   = (const float*)d_in[11];
  const float* fc1w  = (const float*)d_in[12];
  const float* fc1b  = (const float*)d_in[13];
  const float* fc2w  = (const float*)d_in[14];
  const float* fc2b  = (const float*)d_in[15];
  const float* fc3w  = (const float*)d_in[16];
  const float* fc3b  = (const float*)d_in[17];
  float* out = (float*)d_out;

  const int* src = eidx;
  const int* dst = eidx + N_EDGES;

  // ---- workspace layout (float units; all offsets stay 16B aligned) ------
  float* wsf = (float*)d_ws;
  size_t off = 0;
  auto allocF = [&](size_t n) { float* p = wsf + off; off += n; return p; };

  float* T1   = allocF((size_t)N_NODES * D_H1);
  float* A1   = allocF((size_t)N_NODES * D_H1);
  float* H1   = allocF((size_t)N_NODES * D_H1);
  float* T2   = allocF((size_t)N_NODES * D_H2);
  float* A2   = allocF((size_t)N_NODES * D_H2);
  float* H2   = allocF((size_t)N_NODES * D_H2);
  float* deg  = allocF(N_NODES);
  float* dinv = allocF(N_NODES + 48);                // keep 16B alignment
  float* Z1   = allocF((size_t)B_PAIRS * 128);
  float* Z2   = allocF((size_t)B_PAIRS * 32);

  unsigned short* xbf   = (unsigned short*)allocF((size_t)N_NODES * D_IN_P / 2);
  unsigned short* W1t   = (unsigned short*)allocF((size_t)D_H1 * D_IN_P / 2);
  unsigned short* f1wt  = (unsigned short*)allocF((size_t)D_H1 * D_IN_P / 2);
  unsigned short* W2t   = (unsigned short*)allocF((size_t)D_H2 * D_H1 / 2);
  unsigned short* f2wt  = (unsigned short*)allocF((size_t)D_H2 * D_H1 / 2);
  unsigned short* fc1wt = (unsigned short*)allocF((size_t)128 * D_FEAT / 2);
  unsigned short* fc2wt = (unsigned short*)allocF((size_t)32 * 128 / 2);
  // reuse dead f32 regions for later bf16 activations
  unsigned short* H1bf   = (unsigned short*)T1;      // 20.5MB < 41MB
  unsigned short* featbf = (unsigned short*)A1;      // 26.2MB < 41MB
  unsigned short* Z1bf   = (unsigned short*)T2;      // 4.2MB  < 16MB

  auto gemm = [&](const unsigned short* A, const unsigned short* Bt,
                  const float* bias, const float* res, float* C,
                  int M, int N, int Kp, int epi) {
    dim3 g((N + BN - 1) / BN, (M + BM - 1) / BM);
    wmma_gemm_bf16<<<g, 256, 0, stream>>>(A, Bt, bias, res, C, M, N, Kp, epi);
  };
  auto cvt = [&](const float* A, unsigned short* o, int M, int K, int Kp) {
    size_t tot = (size_t)M * Kp;
    convert_pad_bf16<<<(unsigned)((tot + 255) / 256), 256, 0, stream>>>(A, o, M, K, Kp);
  };
  auto tcvt = [&](const float* B, unsigned short* o, int K, int N, int Kp) {
    size_t tot = (size_t)N * Kp;
    transpose_convert_bf16<<<(unsigned)((tot + 255) / 256), 256, 0, stream>>>(B, o, K, N, Kp);
  };

  // ---- one-shot conversions ----------------------------------------------
  cvt(x, xbf, N_NODES, D_IN, D_IN_P);
  tcvt(W1,  W1t,  D_IN,  D_H1, D_IN_P);
  tcvt(f1w, f1wt, D_IN,  D_H1, D_IN_P);
  tcvt(W2,  W2t,  D_H1,  D_H2, D_H1);
  tcvt(f2w, f2wt, D_H1,  D_H2, D_H1);
  tcvt(fc1w, fc1wt, D_FEAT, 128, D_FEAT);
  tcvt(fc2w, fc2wt, 128, 32, 128);

  // ---- degrees & symmetric normalization ---------------------------------
  fill_f32<<<(N_NODES + 255) / 256, 256, 0, stream>>>(deg, 1.0f, N_NODES);
  deg_count<<<(N_EDGES + 255) / 256, 256, 0, stream>>>(dst, deg, N_EDGES);
  rsqrt_k<<<(N_NODES + 255) / 256, 256, 0, stream>>>(deg, dinv, N_NODES);

  // ---- Layer 1 ------------------------------------------------------------
  gemm(xbf, W1t, b1, nullptr, T1, N_NODES, D_H1, D_IN_P, 0);
  {
    size_t tot = (size_t)N_NODES * D_H1;
    self_init<<<(unsigned)((tot + 255) / 256), 256, 0, stream>>>(T1, dinv, A1, N_NODES, D_H1);
    edge_agg<<<N_EDGES, 256, 0, stream>>>(T1, src, dst, dinv, A1, D_H1);
  }
  gemm(xbf, f1wt, f1b, A1, H1, N_NODES, D_H1, D_IN_P, 1);

  // ---- Layer 2 ------------------------------------------------------------
  cvt(H1, H1bf, N_NODES, D_H1, D_H1);               // overlays T1 (dead)
  gemm(H1bf, W2t, b2, nullptr, T2, N_NODES, D_H2, D_H1, 0);
  {
    size_t tot = (size_t)N_NODES * D_H2;
    self_init<<<(unsigned)((tot + 255) / 256), 256, 0, stream>>>(T2, dinv, A2, N_NODES, D_H2);
    edge_agg<<<N_EDGES, 256, 0, stream>>>(T2, src, dst, dinv, A2, D_H2);
  }
  gemm(H1bf, f2wt, f2b, A2, H2, N_NODES, D_H2, D_H1, 1);

  // ---- Pair MLP -----------------------------------------------------------
  {
    size_t tot = (size_t)B_PAIRS * D_FEAT;          // featbf overlays A1 (dead)
    gather_feat_bf16<<<(unsigned)((tot + 255) / 256), 256, 0, stream>>>(H2, pairs, featbf);
  }
  gemm(featbf, fc1wt, fc1b, nullptr, Z1, B_PAIRS, 128, D_FEAT, 1);
  cvt(Z1, Z1bf, B_PAIRS, 128, 128);                 // overlays T2 (dead)
  gemm(Z1bf, fc2wt, fc2b, nullptr, Z2, B_PAIRS, 32, 128, 1);
  fc3_sigmoid<<<(B_PAIRS + 255) / 256, 256, 0, stream>>>(Z2, fc3w, fc3b, out, B_PAIRS);
}
